// NF_regularizer_66614942761139
// MI455X (gfx1250) — matrix operand order repaired and checked by reading, not compiled, not run on hardware
//
#include <hip/hip_runtime.h>
#include <hip/hip_bf16.h>
#include <math.h>

// ---------------- problem constants ----------------
#define NLAYERS 5
#define DIM     36
#define D1      18
#define HID     512
#define BATCH   131072

#define ROWS_PER_WG 64          // 4 waves x 16 rows
#define WG_THREADS  128
#define N_BLOCKS    (BATCH / ROWS_PER_WG)   // 2048

typedef __attribute__((ext_vector_type(16))) _Float16 v16h;
typedef __attribute__((ext_vector_type(8)))  _Float16 v8h;
typedef __attribute__((ext_vector_type(8)))  float    v8f;
typedef __attribute__((ext_vector_type(4)))  int      v4i;

#define AS1 __attribute__((address_space(1)))
#define AS3 __attribute__((address_space(3)))

union HV16 { v16h v; v8h h[2]; };

// CDNA5 async global->LDS path (ASYNCcnt-tracked), with synchronous fallback.
#if defined(__gfx1250__) && __has_builtin(__builtin_amdgcn_global_load_async_to_lds_b128)
#define HAVE_ASYNC_LDS 1
#else
#define HAVE_ASYNC_LDS 0
#endif

__device__ __forceinline__ void wait_async_all() {
#if HAVE_ASYNC_LDS
#if __has_builtin(__builtin_amdgcn_s_wait_asynccnt)
    __builtin_amdgcn_s_wait_asynccnt(0);
#else
    asm volatile("s_wait_asynccnt 0x0" ::: "memory");
#endif
#endif
}

// Copy one 16-output-row W2 slice (16*512 f16 = 16 KB) into LDS.
// Async path: 8 x global_load_async_to_lds_b128 per thread (no VGPR data).
__device__ __forceinline__ void issue_w2_slice(const _Float16* __restrict__ w2h,
                                               _Float16* dst, int nb, int tid) {
    const _Float16* src = w2h + (size_t)nb * 16 * HID;
#if HAVE_ASYNC_LDS
    #pragma unroll
    for (int i = 0; i < 8; ++i) {
        const int e = (tid + i * WG_THREADS) * 8;   // halves (16B chunks)
        AS1 v4i* g = (AS1 v4i*)(uintptr_t)(src + e);           // flat==global addr
        AS3 v4i* l = (AS3 v4i*)(uint32_t)(uintptr_t)(dst + e); // low 32b = LDS offset
        __builtin_amdgcn_global_load_async_to_lds_b128(g, l, 0, 0);
    }
#else
    const uint4* s4 = (const uint4*)src;
    uint4* d4 = (uint4*)dst;
    #pragma unroll
    for (int i = 0; i < 8; ++i) d4[tid + i * WG_THREADS] = s4[tid + i * WG_THREADS];
#endif
}

// ---------------- workspace layout (bytes) ----------------
constexpr size_t N_W1P = 10ull * 512 * 32;       // halves  (K padded 18->32)
constexpr size_t N_W2H = 10ull * 512 * 512;      // halves
constexpr size_t N_W3P = 10ull * 48  * 512;      // halves  (N padded 36->48)
constexpr size_t OFF_W1P  = 0;
constexpr size_t OFF_W2H  = OFF_W1P + N_W1P * 2;
constexpr size_t OFF_W3P  = OFF_W2H + N_W2H * 2;
constexpr size_t OFF_B1   = OFF_W3P + N_W3P * 2;
constexpr size_t OFF_B2   = OFF_B1  + 10ull * 512 * 4;
constexpr size_t OFF_B3   = OFF_B2  + 10ull * 512 * 4;
constexpr size_t OFF_IPERM= OFF_B3  + 10ull * 48 * 4;
constexpr size_t OFF_PART = OFF_IPERM + 5ull * 36 * 4;   // 720B, 16B-multiple

struct ParamPtrs { const float* p[60]; };

// ---------------- prologue: f16 weight conversion w/ padding ----------------
__global__ void nf_prep_weights(ParamPtrs P, _Float16* __restrict__ W1P,
                                _Float16* __restrict__ W2H, _Float16* __restrict__ W3P,
                                float* __restrict__ B1, float* __restrict__ B2,
                                float* __restrict__ B3) {
    const int su = blockIdx.x;          // 0..9 = layer*2 + subnet
    const int t  = threadIdx.x;
    const float* w1 = P.p[su * 6 + 0];
    const float* b1 = P.p[su * 6 + 1];
    const float* w2 = P.p[su * 6 + 2];
    const float* b2 = P.p[su * 6 + 3];
    const float* w3 = P.p[su * 6 + 4];
    const float* b3 = P.p[su * 6 + 5];

    for (int i = t; i < 512 * 32; i += 256) {          // W1 [512,18] -> [512,32]
        int o = i >> 5, kk = i & 31;
        W1P[(size_t)su * 512 * 32 + i] = (_Float16)((kk < D1) ? w1[o * D1 + kk] : 0.f);
    }
    for (int i = t; i < 512 * 512; i += 256)           // W2 [512,512]
        W2H[(size_t)su * 512 * 512 + i] = (_Float16)w2[i];
    for (int i = t; i < 48 * 512; i += 256) {          // W3 [36,512] -> [48,512]
        int o = i >> 9, c = i & 511;
        W3P[(size_t)su * 48 * 512 + i] = (_Float16)((o < DIM) ? w3[o * 512 + c] : 0.f);
    }
    for (int i = t; i < 512; i += 256) { B1[su * 512 + i] = b1[i]; B2[su * 512 + i] = b2[i]; }
    for (int i = t; i < 48;  i += 256) B3[su * 48 + i] = (i < DIM) ? b3[i] : 0.f;
}

// ---------------- prologue: inverse permutations ----------------
__global__ void nf_prep_perm(const int* __restrict__ perms, int* __restrict__ iperm) {
    int i = threadIdx.x;
    if (i < NLAYERS * DIM) {
        int j = i % DIM;
        iperm[(i / DIM) * DIM + perms[i]] = j;   // inv[perm[j]] = j
    }
}

// ---------------- main fused flow kernel ----------------
__global__ __launch_bounds__(WG_THREADS)
void nf_flow_kernel(const float* __restrict__ FD,
                    const _Float16* __restrict__ W1P, const _Float16* __restrict__ W2H,
                    const _Float16* __restrict__ W3P,
                    const float* __restrict__ B1, const float* __restrict__ B2,
                    const float* __restrict__ B3,
                    const int* __restrict__ IPERM, float* __restrict__ partials) {
    __shared__ __align__(16) float    Xb[2][ROWS_PER_WG][40];   // 20480 B (ping-pong x)
    __shared__ __align__(16) _Float16 H1[ROWS_PER_WG][HID];     // 65536 B (hidden 1, f16)
    __shared__ __align__(16) _Float16 Wst[2][16][HID];          // 32768 B (W2 double buffer)
    __shared__ __align__(16) _Float16 Stg[4][16][32];           // 4096 B  (h2 transpose stage)
    __shared__ __align__(16) float    Rb[4][16][48];            // 12288 B (subnet output r)
    __shared__ float Red[WG_THREADS];                           // 512 B

    const int tid   = threadIdx.x;
    const int wave  = tid >> 5;
    const int lane  = tid & 31;
    const int lhalf = lane & 15;
    const int lsel  = lane >> 4;          // 0: lanes 0-15, 1: lanes 16-31
    const int wr0   = wave * 16;          // wave's first local row
    const int rowg0 = blockIdx.x * ROWS_PER_WG;

    float sacc = 0.f;                     // per-lane: sum of s  (+0.5*x^2 at the end)

    // load this block's x tile
    for (int i = tid; i < ROWS_PER_WG * DIM; i += WG_THREADS) {
        int r = i / DIM, c = i % DIM;
        Xb[0][r][c] = FD[(size_t)(rowg0 + r) * DIM + c];
    }
    __syncthreads();

    int pp = 0;
    for (int k = NLAYERS - 1; k >= 0; --k) {
        // ---- inverse permutation (wave-local rows) ----
        const int* ip = IPERM + k * DIM;
        for (int i = lane; i < 16 * DIM; i += 32) {
            int r = i / DIM, c = i % DIM;
            Xb[1 - pp][wr0 + r][c] = Xb[pp][wr0 + r][ip[c]];
        }
        pp ^= 1;
        asm volatile("s_wait_dscnt 0" ::: "memory");

        for (int sub = 0; sub < 2; ++sub) {
            const int su = k * 2 + sub;
            const int inb = (sub == 0) ? 0 : D1;      // subnet input: x1 or y2
            const _Float16* w1p = W1P + (size_t)su * 512 * 32;
            const _Float16* w2h = W2H + (size_t)su * 512 * 512;
            const _Float16* w3p = W3P + (size_t)su * 48 * 512;

            // prefetch W2 slice nb=0 into Wst[0]; layer A hides the latency
            issue_w2_slice(w2h, &Wst[0][0][0], 0, tid);

            // ---- build A-fragment (16x32 f16) from x slice, K padded to 32 ----
            HV16 afx;
            #pragma unroll
            for (int j = 0; j < 8; ++j) {
                int kk = lsel * 8 + j;            // elements 0..7  -> K 0..15
                int k2 = 16 + lsel * 8 + j;       // elements 8..15 -> K 16..31
                afx.v[j]     = (kk < D1) ? (_Float16)Xb[pp][wr0 + lhalf][inb + kk] : (_Float16)0.f;
                afx.v[8 + j] = (k2 < D1) ? (_Float16)Xb[pp][wr0 + lhalf][inb + k2] : (_Float16)0.f;
            }

            // ---- layer A: [16,32] x [32,512] -> H1 (bias+ReLU, f16) ----
            for (int nn = 0; nn < 32; ++nn) {
                const int ncol = nn * 16 + lhalf;
                const _Float16* wp = w1p + (size_t)ncol * 32 + lsel * 8;
                HV16 bf; bf.h[0] = *(const v8h*)wp; bf.h[1] = *(const v8h*)(wp + 16);
                v8f cz = {};
                v8f acc = __builtin_amdgcn_wmma_f32_16x16x32_f16(
                    false, afx.v, false, bf.v, (short)0, cz, false, false);
                const float bias = B1[su * 512 + ncol];
                #pragma unroll
                for (int v = 0; v < 8; ++v) {
                    float xv = fmaxf(acc[v] + bias, 0.f);
                    H1[wr0 + lsel * 8 + v][ncol] = (_Float16)xv;
                }
            }
            asm volatile("s_wait_dscnt 0" ::: "memory");

            // ---- layers B+C fused, double-buffered async W2 staging ----
            v8f accC[3];
            #pragma unroll
            for (int co = 0; co < 3; ++co)
                #pragma unroll
                for (int v = 0; v < 8; ++v) accC[co][v] = 0.f;

            for (int nb = 0; nb < 32; ++nb) {
                const int cur = nb & 1;
                wait_async_all();          // own async slice copies complete in LDS
                __syncthreads();           // publish slice; retire readers of spare buffer
                if (nb < 31)               // stream next slice during the WMMA chain
                    issue_w2_slice(w2h, &Wst[1 - cur][0][0], nb + 1, tid);

                v8f acc = {};
                #pragma unroll 4
                for (int kb = 0; kb < 16; ++kb) {
                    const _Float16* hp = &H1[wr0 + lhalf][kb * 32 + lsel * 8];
                    HV16 af; af.h[0] = *(const v8h*)hp; af.h[1] = *(const v8h*)(hp + 16);
                    const _Float16* bp = &Wst[cur][lhalf][kb * 32 + lsel * 8];
                    HV16 bf; bf.h[0] = *(const v8h*)bp; bf.h[1] = *(const v8h*)(bp + 16);
                    acc = __builtin_amdgcn_wmma_f32_16x16x32_f16(
                        false, af.v, false, bf.v, (short)0, acc, false, false);
                }
                const int ncol = nb * 16 + lhalf;
                const float bias = B2[su * 512 + ncol];
                #pragma unroll
                for (int v = 0; v < 8; ++v) {
                    float xv = fmaxf(acc[v] + bias, 0.f);
                    Stg[wave][lsel * 8 + v][(nb & 1) * 16 + lhalf] = (_Float16)xv;
                }
                if (nb & 1) {   // 32 cols of h2 staged -> feed layer C
                    asm volatile("s_wait_dscnt 0" ::: "memory");
                    const _Float16* sp = &Stg[wave][lhalf][lsel * 8];
                    HV16 af; af.h[0] = *(const v8h*)sp; af.h[1] = *(const v8h*)(sp + 16);
                    const int kc = nb >> 1;
                    #pragma unroll
                    for (int co = 0; co < 3; ++co) {
                        const _Float16* wp = w3p + (size_t)(co * 16 + lhalf) * HID
                                           + kc * 32 + lsel * 8;
                        HV16 bf; bf.h[0] = *(const v8h*)wp; bf.h[1] = *(const v8h*)(wp + 16);
                        accC[co] = __builtin_amdgcn_wmma_f32_16x16x32_f16(
                            false, af.v, false, bf.v, (short)0, accC[co], false, false);
                    }
                }
            }

            // ---- r = accC + b3 -> LDS (fp32) ----
            #pragma unroll
            for (int co = 0; co < 3; ++co) {
                const int col = co * 16 + lhalf;
                const float b3v = B3[su * 48 + col];
                #pragma unroll
                for (int v = 0; v < 8; ++v)
                    Rb[wave][lsel * 8 + v][col] = accC[co][v] + b3v;
            }
            asm volatile("s_wait_dscnt 0" ::: "memory");

            // ---- coupling: s = 1.0176*atan(r[:, :18]); t = r[:, 18:] ----
            for (int i = lane; i < 16 * D1; i += 32) {
                int r = i / D1, j = i % D1;
                float s = 1.0176f * atanf(Rb[wave][r][j]);      // CLAMP*0.636
                float t = Rb[wave][r][D1 + j];
                sacc += s;                                       // reg gets +mean(sum s)
                if (sub == 0) {     // y2 = (x2 - t1)*exp(-s1)  -> cols 18..35
                    float x2 = Xb[pp][wr0 + r][D1 + j];
                    Xb[pp][wr0 + r][D1 + j] = (x2 - t) * expf(-s);
                } else {            // y1 = (x1 - t2)*exp(-s2)  -> cols 0..17
                    float x1 = Xb[pp][wr0 + r][j];
                    Xb[pp][wr0 + r][j] = (x1 - t) * expf(-s);
                }
            }
            asm volatile("s_wait_dscnt 0" ::: "memory");
        }
    }

    // ---- 0.5*||x||^2 and block reduction ----
    for (int i = lane; i < 16 * DIM; i += 32) {
        int r = i / DIM, c = i % DIM;
        float v = Xb[pp][wr0 + r][c];
        sacc += 0.5f * v * v;
    }
    Red[tid] = sacc;
    __syncthreads();
    for (int off = WG_THREADS / 2; off > 0; off >>= 1) {
        if (tid < off) Red[tid] += Red[tid + off];
        __syncthreads();
    }
    if (tid == 0) partials[blockIdx.x] = Red[0];
}

// ---------------- final deterministic reduction ----------------
__global__ void nf_reduce(const float* __restrict__ part, float* __restrict__ out, int n) {
    __shared__ float red[256];
    float s = 0.f;
    for (int i = threadIdx.x; i < n; i += 256) s += part[i];
    red[threadIdx.x] = s;
    __syncthreads();
    for (int off = 128; off > 0; off >>= 1) {
        if (threadIdx.x < off) red[threadIdx.x] += red[threadIdx.x + off];
        __syncthreads();
    }
    if (threadIdx.x == 0) out[0] = red[0] / (float)BATCH;
}

// ---------------- launch ----------------
extern "C" void kernel_launch(void* const* d_in, const int* in_sizes, int n_in,
                              void* d_out, int out_size, void* d_ws, size_t ws_size,
                              hipStream_t stream) {
    const float* fd    = (const float*)d_in[0];
    const int*   perms = (const int*)d_in[1 + 60];

    char* ws = (char*)d_ws;
    _Float16* W1P = (_Float16*)(ws + OFF_W1P);
    _Float16* W2H = (_Float16*)(ws + OFF_W2H);
    _Float16* W3P = (_Float16*)(ws + OFF_W3P);
    float* B1 = (float*)(ws + OFF_B1);
    float* B2 = (float*)(ws + OFF_B2);
    float* B3 = (float*)(ws + OFF_B3);
    int*   IP = (int*)(ws + OFF_IPERM);
    float* PART = (float*)(ws + OFF_PART);

    ParamPtrs P;
    for (int m = 0; m < 60; ++m) P.p[m] = (const float*)d_in[1 + m];

    nf_prep_perm<<<1, 256, 0, stream>>>(perms, IP);
    nf_prep_weights<<<10, 256, 0, stream>>>(P, W1P, W2H, W3P, B1, B2, B3);
    nf_flow_kernel<<<N_BLOCKS, WG_THREADS, 0, stream>>>(fd, W1P, W2H, W3P, B1, B2, B3, IP, PART);
    nf_reduce<<<1, 256, 0, stream>>>(PART, (float*)d_out, N_BLOCKS);
}
